// MultiHeadAttention_33097017983531
// MI455X (gfx1250) — matrix operand order repaired and checked by reading, not compiled, not run on hardware
//
#include <hip/hip_runtime.h>
#include <hip/hip_bf16.h>

// ---------------- problem constants ----------------
constexpr int B  = 4;
constexpr int S  = 1024;
constexpr int D  = 1024;
constexpr int H  = 16;
constexpr int HD = 64;          // head dim
constexpr int BS = B * S;       // 4096 rows
constexpr float EPS   = 1e-6f;
constexpr float LOG2E = 1.442695041f;

// ---------------- vector types ----------------
typedef __attribute__((ext_vector_type(16))) __bf16 v16bf;
typedef __attribute__((ext_vector_type(8)))  __bf16 v8bf;
typedef __attribute__((ext_vector_type(8)))  float  v8f;
typedef __attribute__((ext_vector_type(4)))  int    v4i;

__device__ __forceinline__ v16bf cat8(v8bf lo, v8bf hi) {
  return __builtin_shufflevector(lo, hi, 0,1,2,3,4,5,6,7,8,9,10,11,12,13,14,15);
}

__device__ __forceinline__ v8f wmma_bf16(v16bf a, v16bf b, v8f c) {
  // D = A(16x32 bf16) * B(32x16 bf16) + C(16x16 f32)
  return __builtin_amdgcn_wmma_f32_16x16x32_bf16(
      /*neg_a=*/false, a, /*neg_b=*/false, b,
      /*c_mod=*/(short)0, c, /*reuse_a=*/false, /*reuse_b=*/false);
}

// ---------------- async global->LDS helpers (guarded) ----------------
__device__ __forceinline__ void async_ld16(__bf16* lds, const __bf16* g) {
#if __has_builtin(__builtin_amdgcn_global_load_async_to_lds_b128)
  // probe-learned signature: (int4 addrspace(1)*, addrspace(3) dst, imm, imm)
  __builtin_amdgcn_global_load_async_to_lds_b128(
      (__attribute__((address_space(1))) v4i*)g,
      (__attribute__((address_space(3))) v4i*)lds, 0, 0);
#else
  *(v8bf*)lds = *(const v8bf*)g;  // fallback: stage through VGPRs
#endif
}

__device__ __forceinline__ void wait_async() {
#if __has_builtin(__builtin_amdgcn_s_wait_asynccnt)
  __builtin_amdgcn_s_wait_asynccnt(0);
#else
  asm volatile("s_wait_asynccnt 0" ::: "memory");
#endif
}

// lane-half (16-lane) reductions: C-layout keeps row r in lanes 0-15 and
// row r+8 in lanes 16-31, so xor masks 1..8 never cross halves.
__device__ __forceinline__ float half16_max(float x) {
  #pragma unroll
  for (int m = 1; m <= 8; m <<= 1) x = fmaxf(x, __shfl_xor(x, m, 32));
  return x;
}
__device__ __forceinline__ float half16_sum(float x) {
  #pragma unroll
  for (int m = 1; m <= 8; m <<= 1) x += __shfl_xor(x, m, 32);
  return x;
}
__device__ __forceinline__ float wave_sum(float x) {
  #pragma unroll
  for (int m = 1; m <= 16; m <<= 1) x += __shfl_xor(x, m, 32);
  return x;
}

// ---------------- kernels ----------------

__global__ void f32_to_bf16_k(const float* __restrict__ x,
                              __bf16* __restrict__ y, int n) {
  for (int i = blockIdx.x * blockDim.x + threadIdx.x; i < n;
       i += gridDim.x * blockDim.x)
    y[i] = (__bf16)x[i];
}

__global__ void mask_count_k(const unsigned char* __restrict__ pm,
                             int* __restrict__ nm) {
  int b = threadIdx.x;
  if (b < B) {
    int s = 0;
    for (int i = 0; i < S; ++i) s += (pm[b * S + i] != 0);
    nm[b] = s;
  }
}

// C[M,N](f32) = A[M,K](bf16) * W[N,K]^T(bf16).
// Block tile 128x128, 8 waves (4 along M x 2 along N), wave tile 32x64.
// K stepped by 32 with double-buffered async global->LDS staging.
// LDS row stride padded to 40 elems (80B): ds_load_b128 fragment fetches hit
// distinct bank windows (20*r mod 64 injective for r=0..15).
__global__ __launch_bounds__(256)
void gemm_bf16_f32_k(const __bf16* __restrict__ A,
                     const __bf16* __restrict__ W,
                     float* __restrict__ C, int M, int N, int K) {
  constexpr int LDP = 40;  // padded LDS row stride (elements)
  __shared__ __bf16 lA[2][128 * LDP];
  __shared__ __bf16 lB[2][128 * LDP];

  const int lane = threadIdx.x & 31;
  const int wave = threadIdx.x >> 5;
  const int wm = wave & 3;          // 4 waves along M
  const int wn = wave >> 2;         // 2 waves along N
  const int m0b = blockIdx.y * 128;
  const int n0b = blockIdx.x * 128;
  const int m0 = m0b + wm * 32;
  const int n0 = n0b + wn * 64;
  const int l16 = lane & 15;
  const int lh  = lane >> 4;

  // cooperative staging: 128 rows x 32 cols (64B/row) per matrix = 512
  // 16B-chunks; 2 chunks per thread per matrix.
  auto stage = [&](int bsel, int kk) {
    #pragma unroll
    for (int u = 0; u < 2; ++u) {
      const int ch  = u * 256 + (int)threadIdx.x;
      const int row = ch >> 2;
      const int co  = (ch & 3) << 3;
      async_ld16(&lA[bsel][row * LDP + co],
                 A + (size_t)(m0b + row) * K + kk + co);
      async_ld16(&lB[bsel][row * LDP + co],
                 W + (size_t)(n0b + row) * K + kk + co);
    }
  };

  v8f acc[2][4];
  #pragma unroll
  for (int i = 0; i < 2; ++i)
    #pragma unroll
    for (int j = 0; j < 4; ++j) acc[i][j] = {};

  stage(0, 0);
  wait_async();
  __syncthreads();

  int buf = 0;
  for (int kk = 0; kk < K; kk += 32) {
    if (kk + 32 < K) stage(buf ^ 1, kk + 32);  // uniform branch

    // A fragments (16x32): lanes<16 hold K {0..7,16..23}, lanes>=16 {8..15,24..31}
    v16bf a[2];
    #pragma unroll
    for (int i = 0; i < 2; ++i) {
      const __bf16* ap = &lA[buf][(wm * 32 + i * 16 + l16) * LDP + lh * 8];
      a[i] = cat8(*(const v8bf*)ap, *(const v8bf*)(ap + 16));
    }
    // B fragments (32x16): lanes<16 hold K {0..7,16..23}, lanes>=16 {8..15,24..31}
    v16bf bfr[4];
    #pragma unroll
    for (int j = 0; j < 4; ++j) {
      const __bf16* bp = &lB[buf][(wn * 64 + j * 16 + l16) * LDP + lh * 8];
      bfr[j] = cat8(*(const v8bf*)bp, *(const v8bf*)(bp + 16));
    }
    #pragma unroll
    for (int i = 0; i < 2; ++i)
      #pragma unroll
      for (int j = 0; j < 4; ++j)
        acc[i][j] = wmma_bf16(a[i], bfr[j], acc[i][j]);

    wait_async();
    __syncthreads();
    buf ^= 1;
  }

  // C layout: VGPR r -> row m0+r (lanes 0-15) / m0+8+r (lanes 16-31), col n0+l16
  #pragma unroll
  for (int i = 0; i < 2; ++i)
    #pragma unroll
    for (int j = 0; j < 4; ++j)
      #pragma unroll
      for (int r = 0; r < 8; ++r)
        C[(size_t)(m0 + i * 16 + r + 8 * lh) * N + n0 + j * 16 + l16] =
            acc[i][j][r];
}

// fused RoPE + RMSNorm + per-dim softplus scale; emits attention-friendly
// layouts: Qbf/Kbf [B,H,S,HD] row-major, Vt [B,H,HD,S] (pre-transposed).
__global__ __launch_bounds__(256)
void qkv_pack_k(const float* __restrict__ Qf, const float* __restrict__ Kf,
                const float* __restrict__ Vf, const float* __restrict__ qln,
                const float* __restrict__ kln, const float* __restrict__ pds,
                const int* __restrict__ nm, __bf16* __restrict__ Qbf,
                __bf16* __restrict__ Kbf, __bf16* __restrict__ Vt) {
  const int lane = threadIdx.x & 31;
  const int row  = blockIdx.x * 8 + (threadIdx.x >> 5);  // (b*S + s)*H + h
  const int h  = row % H;
  const int bs = row / H;
  const int s  = bs % S;
  const int b  = bs / S;
  const size_t base = (size_t)bs * D + h * HD;  // [b,s,h,:]

  float x1q = Qf[base + lane], x2q = Qf[base + lane + 32];
  float x1k = Kf[base + lane], x2k = Kf[base + lane + 32];

  // RoPE: lane l rotates the pair (l, l+32); 1/ts = exp(-l*ln(10000)/32)
  float pos = (float)s - (float)nm[b];
  float inv_ts = __expf(-0.28782313662425287f * (float)lane);
  float sn, cs;
  __sincosf(pos * inv_ts, &sn, &cs);
  float q1 = x1q * cs - x2q * sn, q2 = x2q * cs + x1q * sn;
  float k1 = x1k * cs - x2k * sn, k2 = x2k * cs + x1k * sn;

  // RMSNorm over the 64 dims of this head row
  float rq = rsqrtf(wave_sum(q1 * q1 + q2 * q2) * (1.0f / HD) + EPS);
  float rk = rsqrtf(wave_sum(k1 * k1 + k2 * k2) * (1.0f / HD) + EPS);

  auto softplus = [](float x) { return x > 15.0f ? x : log1pf(__expf(x)); };
  const float qs = LOG2E * 0.125f;  // LOG2E / sqrt(HD)
  float sc1 = qs * softplus(pds[lane]);
  float sc2 = qs * softplus(pds[lane + 32]);

  q1 = q1 * rq * qln[lane] * sc1;
  q2 = q2 * rq * qln[lane + 32] * sc2;
  k1 = k1 * rk * kln[lane];
  k2 = k2 * rk * kln[lane + 32];

  const size_t ob = (((size_t)b * H + h) * S + s) * HD;
  Qbf[ob + lane]      = (__bf16)q1;
  Qbf[ob + lane + 32] = (__bf16)q2;
  Kbf[ob + lane]      = (__bf16)k1;
  Kbf[ob + lane + 32] = (__bf16)k2;

  const size_t vb = ((size_t)b * H + h) * HD * S + s;  // [b,h,d,s]
  Vt[vb + (size_t)lane * S]        = (__bf16)Vf[base + lane];
  Vt[vb + (size_t)(lane + 32) * S] = (__bf16)Vf[base + lane + 32];
}

// flash attention: one wave per 16-row q-tile, online softmax over 32-wide
// causal k-blocks.  4 independent waves per block (own LDS P buffers).
__global__ __launch_bounds__(128)
void flash_attn_k(const __bf16* __restrict__ Qbf, const __bf16* __restrict__ Kbf,
                  const __bf16* __restrict__ Vt, const int* __restrict__ nm,
                  __bf16* __restrict__ Xatt) {
  __shared__ __bf16 Pt[4][16 * 32];
  const int lane = threadIdx.x & 31;
  const int wave = threadIdx.x >> 5;
  const int l16 = lane & 15;
  const int lh  = lane >> 4;
  const int q0 = blockIdx.x * 64 + wave * 16;
  const int h  = blockIdx.y;
  const int b  = blockIdx.z;
  const int nmb = nm[b];
  __bf16* pt = Pt[wave];

  const __bf16* Qh = Qbf + ((size_t)b * H + h) * S * HD;
  const __bf16* Kh = Kbf + ((size_t)b * H + h) * S * HD;
  const __bf16* Vh = Vt  + ((size_t)b * H + h) * HD * S;

  // Q A-fragments: 2 K-chunks of 32 over d
  v16bf qf[2];
  #pragma unroll
  for (int dt = 0; dt < 2; ++dt) {
    const __bf16* qp = Qh + (size_t)(q0 + l16) * HD + dt * 32 + lh * 16;
    qf[dt] = cat8(*(const v8bf*)qp, *(const v8bf*)(qp + 8));
  }

  float m_i[8], l_i[8];
  v8f o[4];
  #pragma unroll
  for (int r = 0; r < 8; ++r) { m_i[r] = -1e30f; l_i[r] = 0.0f; }
  #pragma unroll
  for (int j = 0; j < 4; ++j) o[j] = {};

  for (int kb = 0; kb <= q0; kb += 32) {
    // K B-fragments [k-subtile][d-chunk]: column = Kbf row (contiguous d)
    v16bf kf[2][2];
    #pragma unroll
    for (int t = 0; t < 2; ++t)
      #pragma unroll
      for (int dt = 0; dt < 2; ++dt)
        kf[t][dt] = *(const v16bf*)(Kh + (size_t)(kb + t * 16 + l16) * HD +
                                    dt * 32 + lh * 16);

    v8f s0 = {}, s1 = {};
    s0 = wmma_bf16(qf[0], kf[0][0], s0);
    s0 = wmma_bf16(qf[1], kf[0][1], s0);
    s1 = wmma_bf16(qf[0], kf[1][0], s1);
    s1 = wmma_bf16(qf[1], kf[1][1], s1);

    // mask + online softmax (per C-layout row r + 8*lh)
    #pragma unroll
    for (int r = 0; r < 8; ++r) {
      const int mrow = q0 + r + 8 * lh;
      const int c0 = kb + l16, c1 = kb + 16 + l16;
      float v0 = (mrow >= c0 && c0 >= nmb) ? s0[r] : -1e30f;
      float v1 = (mrow >= c1 && c1 >= nmb) ? s1[r] : -1e30f;
      float mnew = fmaxf(m_i[r], half16_max(fmaxf(v0, v1)));
      float alpha = __expf(m_i[r] - mnew);
      m_i[r] = mnew;
      float p0 = __expf(v0 - mnew);
      float p1 = __expf(v1 - mnew);
      l_i[r] = l_i[r] * alpha + half16_sum(p0 + p1);
      #pragma unroll
      for (int j = 0; j < 4; ++j) o[j][r] *= alpha;
      const int rr = r + 8 * lh;
      pt[rr * 32 + l16]      = (__bf16)p0;
      pt[rr * 32 + 16 + l16] = (__bf16)p1;
    }

    // same-wave LDS is in-order; explicit wait + compiler fence before reload
    asm volatile("s_wait_dscnt 0" ::: "memory");

    // P C-layout -> A-layout through LDS
    const __bf16* pp = pt + l16 * 32 + lh * 16;
    v16bf pf = cat8(*(const v8bf*)pp, *(const v8bf*)(pp + 8));

    // V B-fragments: column d = Vt row (contiguous k)
    #pragma unroll
    for (int j = 0; j < 4; ++j) {
      v16bf vf = *(const v16bf*)(Vh + (size_t)(j * 16 + l16) * S + kb + lh * 16);
      o[j] = wmma_bf16(pf, vf, o[j]);
    }
  }

  // normalize and write [b, s, h*HD + d] as bf16 for the output projection
  #pragma unroll
  for (int r = 0; r < 8; ++r) {
    float inv = 1.0f / l_i[r];
    #pragma unroll
    for (int j = 0; j < 4; ++j)
      Xatt[(size_t)(b * S + q0 + r + 8 * lh) * D + h * HD + j * 16 + l16] =
          (__bf16)(o[j][r] * inv);
  }
}

// ---------------- workspace layout (bytes) ----------------
constexpr size_t MB = 1u << 20;
constexpr size_t OFF_XBF  = 0;        // 8 MB  X bf16 [BS,D]
constexpr size_t OFF_WQ   = 8 * MB;   // 2 MB each
constexpr size_t OFF_WK   = 10 * MB;
constexpr size_t OFF_WV   = 12 * MB;
constexpr size_t OFF_WO   = 14 * MB;
constexpr size_t OFF_QF   = 16 * MB;  // 16 MB each (fp32 QKV)
constexpr size_t OFF_KF   = 32 * MB;
constexpr size_t OFF_VF   = 48 * MB;
constexpr size_t OFF_QBF  = 64 * MB;  // 8 MB each
constexpr size_t OFF_KBF  = 72 * MB;
constexpr size_t OFF_VT   = 80 * MB;
constexpr size_t OFF_XATT = 88 * MB;  // 8 MB
constexpr size_t OFF_NM   = 96 * MB;  // 16 B

extern "C" void kernel_launch(void* const* d_in, const int* in_sizes, int n_in,
                              void* d_out, int out_size, void* d_ws,
                              size_t ws_size, hipStream_t stream) {
  const float* x  = (const float*)d_in[0];
  const unsigned char* pm = (const unsigned char*)d_in[1];
  const float* Wq = (const float*)d_in[2];
  const float* Wk = (const float*)d_in[3];
  const float* Wv = (const float*)d_in[4];
  const float* Wo = (const float*)d_in[5];
  const float* qln = (const float*)d_in[6];
  const float* kln = (const float*)d_in[7];
  const float* pds = (const float*)d_in[8];
  float* out = (float*)d_out;
  char* ws = (char*)d_ws;

  __bf16* Xbf  = (__bf16*)(ws + OFF_XBF);
  __bf16* Wqb  = (__bf16*)(ws + OFF_WQ);
  __bf16* Wkb  = (__bf16*)(ws + OFF_WK);
  __bf16* Wvb  = (__bf16*)(ws + OFF_WV);
  __bf16* Wob  = (__bf16*)(ws + OFF_WO);
  float*  Qf   = (float*)(ws + OFF_QF);
  float*  Kf   = (float*)(ws + OFF_KF);
  float*  Vf   = (float*)(ws + OFF_VF);
  __bf16* Qbf  = (__bf16*)(ws + OFF_QBF);
  __bf16* Kbf  = (__bf16*)(ws + OFF_KBF);
  __bf16* Vt   = (__bf16*)(ws + OFF_VT);
  __bf16* Xatt = (__bf16*)(ws + OFF_XATT);
  int*    nmp  = (int*)(ws + OFF_NM);

  f32_to_bf16_k<<<1024, 256, 0, stream>>>(x, Xbf, BS * D);
  f32_to_bf16_k<<<512, 256, 0, stream>>>(Wq, Wqb, D * D);
  f32_to_bf16_k<<<512, 256, 0, stream>>>(Wk, Wkb, D * D);
  f32_to_bf16_k<<<512, 256, 0, stream>>>(Wv, Wvb, D * D);
  f32_to_bf16_k<<<512, 256, 0, stream>>>(Wo, Wob, D * D);
  mask_count_k<<<1, 32, 0, stream>>>(pm, nmp);

  dim3 ggrid(D / 128, BS / 128);
  gemm_bf16_f32_k<<<ggrid, 256, 0, stream>>>(Xbf, Wqb, Qf, BS, D, D);
  gemm_bf16_f32_k<<<ggrid, 256, 0, stream>>>(Xbf, Wkb, Kf, BS, D, D);
  gemm_bf16_f32_k<<<ggrid, 256, 0, stream>>>(Xbf, Wvb, Vf, BS, D, D);

  qkv_pack_k<<<(B * S * H) / 8, 256, 0, stream>>>(Qf, Kf, Vf, qln, kln, pds,
                                                  nmp, Qbf, Kbf, Vt);

  flash_attn_k<<<dim3(S / 64, H, B), 128, 0, stream>>>(Qbf, Kbf, Vt, nmp, Xatt);

  gemm_bf16_f32_k<<<ggrid, 256, 0, stream>>>(Xatt, Wob, out, BS, D, D);
}
